// RGCNBlock_26036091748512
// MI455X (gfx1250) — compile-verified, matrix operand order checked by reading, not run on hardware
//
#include <hip/hip_runtime.h>
#include <stdint.h>

// ---------------------------------------------------------------------------
// RGCN block (2 layers) for MI455X / gfx1250, wave32.
//   Y = gelu(LN(x));  H[N,1152] = Y @ [W0..W7 | root]  (bf16x3 WMMA, f32 acc)
//   agg = root_part + bias (+identity on layer2);  edges: agg[dst] += sum_r attr_r*H[src,r]
// ---------------------------------------------------------------------------

#define N_NODES 50000
#define N_EDGES 600000
#define DIM     128
#define NREL    8
#define NCOLS   (NREL * DIM + DIM)   // 1152 = 8 relations + root
#define LN_EPS  1e-5f

typedef __attribute__((ext_vector_type(16))) __bf16 v16bf;
typedef __attribute__((ext_vector_type(8)))  float  v8f;

union Frag { v16bf v; uint4 q[2]; };   // 32 bytes: 16 bf16 lanes-worth of A/B fragment

__device__ __forceinline__ unsigned short f32_to_bf16(float f) {
  union { float f; unsigned u; } c; c.f = f;
  unsigned u = c.u;
  unsigned r = u + 0x7FFFu + ((u >> 16) & 1u);   // round to nearest even
  return (unsigned short)(r >> 16);
}
__device__ __forceinline__ float bf16_to_f32(unsigned short h) {
  union { unsigned u; float f; } c; c.u = ((unsigned)h) << 16;
  return c.f;
}

// ---------------------------------------------------------------------------
// Pack [W(8x128x128) | root(128x128)] into transposed bf16 hi/lo: BT[col][k].
// col < 1024: col = r*128+j  ->  W[r][k][j];  col >= 1024: root[k][col-1024].
// ---------------------------------------------------------------------------
__global__ __launch_bounds__(256) void pack_weights(
    const float* __restrict__ W, const float* __restrict__ root,
    unsigned short* __restrict__ bthi, unsigned short* __restrict__ btlo) {
  int idx = blockIdx.x * 256 + threadIdx.x;          // col*128 + k
  if (idx >= NCOLS * DIM) return;
  int col = idx >> 7, k = idx & (DIM - 1);
  float w;
  if (col < NREL * DIM) {
    int r = col >> 7, j = col & (DIM - 1);
    w = W[((size_t)r * DIM + k) * DIM + j];
  } else {
    w = root[(size_t)k * DIM + (col - NREL * DIM)];
  }
  unsigned short hi = f32_to_bf16(w);
  bthi[idx] = hi;
  btlo[idx] = f32_to_bf16(w - bf16_to_f32(hi));
}

// ---------------------------------------------------------------------------
// Fused LayerNorm + exact GELU, emit bf16 hi/lo split. One wave per row,
// 4 elements per lane, wave32 shuffle reductions.
// ---------------------------------------------------------------------------
__global__ __launch_bounds__(256) void ln_gelu_split(
    const float* __restrict__ x, const float* __restrict__ g,
    const float* __restrict__ b,
    unsigned short* __restrict__ yhi, unsigned short* __restrict__ ylo) {
  int lane = threadIdx.x & 31;
  int row  = blockIdx.x * 8 + (threadIdx.x >> 5);
  if (row >= N_NODES) return;

  const float4 xv = *(const float4*)&x[(size_t)row * DIM + lane * 4];
  float s = xv.x + xv.y + xv.z + xv.w;
  #pragma unroll
  for (int o = 16; o > 0; o >>= 1) s += __shfl_xor(s, o, 32);
  float mu = s * (1.0f / DIM);

  float d0 = xv.x - mu, d1 = xv.y - mu, d2 = xv.z - mu, d3 = xv.w - mu;
  float v = d0 * d0 + d1 * d1 + d2 * d2 + d3 * d3;
  #pragma unroll
  for (int o = 16; o > 0; o >>= 1) v += __shfl_xor(v, o, 32);
  float rstd = rsqrtf(v * (1.0f / DIM) + LN_EPS);

  const float4 gv = *(const float4*)&g[lane * 4];
  const float4 bv = *(const float4*)&b[lane * 4];
  float yv[4] = { d0 * rstd * gv.x + bv.x, d1 * rstd * gv.y + bv.y,
                  d2 * rstd * gv.z + bv.z, d3 * rstd * gv.w + bv.w };

  ushort4 h, l;
  unsigned short* hp = &h.x;
  unsigned short* lp = &l.x;
  #pragma unroll
  for (int i = 0; i < 4; ++i) {
    float y = yv[i];
    float ge = 0.5f * y * (1.0f + erff(y * 0.70710678118654752f));  // exact GELU
    unsigned short hi = f32_to_bf16(ge);
    hp[i] = hi;
    lp[i] = f32_to_bf16(ge - bf16_to_f32(hi));
  }
  *(ushort4*)&yhi[(size_t)row * DIM + lane * 4] = h;
  *(ushort4*)&ylo[(size_t)row * DIM + lane * 4] = l;
}

// ---------------------------------------------------------------------------
// WMMA GEMM: H[N x 1152] = Y[N x 128] @ BT^T, bf16x3 (hi*hi + hi*lo + lo*hi),
// f32 accumulation. One 16x16 output tile per wave; 4 waves/block cover 4
// consecutive column tiles of one row tile. K = 128 -> 4 k-steps of 32.
//
// A fragment (16-bit A 16x32, ISA 7.12.2): lane l<16 -> row l, K {0..7,16..23};
// lane l>=16 -> row l-16, K {8..15,24..31}. Two contiguous 16B loads per frag.
// B fragment mirrors A with columns (BT is pre-transposed so loads match).
// C/D: lane -> column (l&15); VGPR j -> row (l<16 ? j : 8+j).
// ---------------------------------------------------------------------------
__global__ __launch_bounds__(128) void gemm_wmma(
    const unsigned short* __restrict__ yhi, const unsigned short* __restrict__ ylo,
    const unsigned short* __restrict__ bthi, const unsigned short* __restrict__ btlo,
    float* __restrict__ H) {
  int lane = threadIdx.x & 31;
  int wave = threadIdx.x >> 5;
  int row0 = blockIdx.x * 16;                 // 3125 row tiles * 16 = 50000
  int col0 = (blockIdx.y * 4 + wave) * 16;    // 18*4 = 72 col tiles * 16 = 1152
  int half = lane >> 4;
  int m = row0 + (lane & 15);
  int n = col0 + (lane & 15);

  v8f c = {};
  #pragma unroll
  for (int kt = 0; kt < 4; ++kt) {
    int bk = kt * 32 + half * 8;
    Frag ah, al, bh, bl;
    ah.q[0] = *(const uint4*)&yhi[(size_t)m * DIM + bk];
    ah.q[1] = *(const uint4*)&yhi[(size_t)m * DIM + bk + 16];
    al.q[0] = *(const uint4*)&ylo[(size_t)m * DIM + bk];
    al.q[1] = *(const uint4*)&ylo[(size_t)m * DIM + bk + 16];
    bh.q[0] = *(const uint4*)&bthi[(size_t)n * DIM + bk];
    bh.q[1] = *(const uint4*)&bthi[(size_t)n * DIM + bk + 16];
    bl.q[0] = *(const uint4*)&btlo[(size_t)n * DIM + bk];
    bl.q[1] = *(const uint4*)&btlo[(size_t)n * DIM + bk + 16];
    // bf16x3 compensated product ~ fp32 accuracy at 3 WMMA per k-step
    c = __builtin_amdgcn_wmma_f32_16x16x32_bf16(false, ah.v, false, bh.v, (short)0, c, false, false);
    c = __builtin_amdgcn_wmma_f32_16x16x32_bf16(false, ah.v, false, bl.v, (short)0, c, false, false);
    c = __builtin_amdgcn_wmma_f32_16x16x32_bf16(false, al.v, false, bh.v, (short)0, c, false, false);
  }

  int mbase = row0 + half * 8;
  #pragma unroll
  for (int j = 0; j < 8; ++j)
    H[(size_t)(mbase + j) * NCOLS + (size_t)col0 + (lane & 15)] = c[j];
}

// ---------------------------------------------------------------------------
// agg[n][j] = H[n][1024+j] (root part) + bias[j] (+ identity[n][j] on layer 2)
// Fully overwrites agg, so no memset of poisoned buffers needed.
// ---------------------------------------------------------------------------
__global__ __launch_bounds__(256) void init_agg(
    const float* __restrict__ H, const float* __restrict__ bias,
    const float* __restrict__ ident, float* __restrict__ agg) {
  int idx = blockIdx.x * 256 + threadIdx.x;
  if (idx >= N_NODES * DIM) return;
  int nrow = idx >> 7, j = idx & (DIM - 1);
  float v = H[(size_t)nrow * NCOLS + NREL * DIM + j] + bias[j];
  if (ident) v += ident[idx];
  agg[idx] = v;
}

// ---------------------------------------------------------------------------
// Edge phase: one wave per edge. Lane covers 4 of 128 channels.
// msg = sum_r attr[e][r] * H[src][r*128 + c], scatter-add into agg[dst].
// Gathers hit L2 (H working set ~230MB, avg degree 12); atomics resolve in L2.
// ---------------------------------------------------------------------------
__global__ __launch_bounds__(256) void edge_scatter(
    const float* __restrict__ H, const long long* __restrict__ ei,
    const float* __restrict__ attr, float* __restrict__ agg) {
  int lane = threadIdx.x & 31;
  int e = blockIdx.x * 8 + (threadIdx.x >> 5);
  if (e >= N_EDGES) return;

  int src = (int)ei[e];
  int dst = (int)ei[(size_t)N_EDGES + e];
  float av = (lane < NREL) ? attr[(size_t)e * NREL + lane] : 0.0f;

  const float* hs = H + (size_t)src * NCOLS + lane * 4;
  float4 acc = {0.f, 0.f, 0.f, 0.f};
  #pragma unroll
  for (int r = 0; r < NREL; ++r) {
    float a = __shfl(av, r, 32);
    float4 h = *(const float4*)(hs + r * DIM);
    acc.x += a * h.x; acc.y += a * h.y; acc.z += a * h.z; acc.w += a * h.w;
  }
  float* out = agg + (size_t)dst * DIM + lane * 4;
  unsafeAtomicAdd(out + 0, acc.x);   // -> global_atomic_add_f32 (L2-resident)
  unsafeAtomicAdd(out + 1, acc.y);
  unsafeAtomicAdd(out + 2, acc.z);
  unsafeAtomicAdd(out + 3, acc.w);
}

// ---------------------------------------------------------------------------
extern "C" void kernel_launch(void* const* d_in, const int* in_sizes, int n_in,
                              void* d_out, int out_size, void* d_ws, size_t ws_size,
                              hipStream_t stream) {
  const float*     x     = (const float*)d_in[0];
  const long long* ei    = (const long long*)d_in[1];   // int64 [2, E]
  const float*     attr  = (const float*)d_in[2];
  const float*     ln1g  = (const float*)d_in[3];
  const float*     ln1b  = (const float*)d_in[4];
  const float*     W1    = (const float*)d_in[5];
  const float*     root1 = (const float*)d_in[6];
  const float*     b1    = (const float*)d_in[7];
  const float*     ln2g  = (const float*)d_in[8];
  const float*     ln2b  = (const float*)d_in[9];
  const float*     W2    = (const float*)d_in[10];
  const float*     root2 = (const float*)d_in[11];
  const float*     b2    = (const float*)d_in[12];
  float* out = (float*)d_out;

  // Workspace carve-up (256B aligned)
  char* ws = (char*)d_ws;
  size_t off = 0;
  auto take = [&](size_t bytes) { void* p = ws + off; off = (off + bytes + 255) & ~(size_t)255; return p; };
  const size_t BT_ELTS = (size_t)NCOLS * DIM;
  unsigned short* bt1h = (unsigned short*)take(BT_ELTS * 2);
  unsigned short* bt1l = (unsigned short*)take(BT_ELTS * 2);
  unsigned short* bt2h = (unsigned short*)take(BT_ELTS * 2);
  unsigned short* bt2l = (unsigned short*)take(BT_ELTS * 2);
  unsigned short* yhi  = (unsigned short*)take((size_t)N_NODES * DIM * 2);
  unsigned short* ylo  = (unsigned short*)take((size_t)N_NODES * DIM * 2);
  float*          agg1 = (float*)take((size_t)N_NODES * DIM * 4);
  float*          H    = (float*)take((size_t)N_NODES * NCOLS * 4);   // ~230 MB
  (void)ws_size; (void)n_in; (void)in_sizes; (void)out_size;

  const int packBlocks = (NCOLS * DIM + 255) / 256;        // 576
  const int lnBlocks   = (N_NODES + 7) / 8;                // 6250
  const dim3 gemmGrid(N_NODES / 16, NCOLS / (16 * 4));     // (3125, 18)
  const int aggBlocks  = (N_NODES * DIM + 255) / 256;      // 25000
  const int edgeBlocks = (N_EDGES + 7) / 8;                // 75000

  pack_weights<<<packBlocks, 256, 0, stream>>>(W1, root1, bt1h, bt1l);
  pack_weights<<<packBlocks, 256, 0, stream>>>(W2, root2, bt2h, bt2l);

  // ---- layer 1 ----
  ln_gelu_split<<<lnBlocks, 256, 0, stream>>>(x, ln1g, ln1b, yhi, ylo);
  gemm_wmma<<<gemmGrid, 128, 0, stream>>>(yhi, ylo, bt1h, bt1l, H);
  init_agg<<<aggBlocks, 256, 0, stream>>>(H, b1, nullptr, agg1);
  edge_scatter<<<edgeBlocks, 256, 0, stream>>>(H, ei, attr, agg1);

  // ---- layer 2 (identity folded into init) ----
  ln_gelu_split<<<lnBlocks, 256, 0, stream>>>(agg1, ln2g, ln2b, yhi, ylo);
  gemm_wmma<<<gemmGrid, 128, 0, stream>>>(yhi, ylo, bt2h, bt2l, H);
  init_agg<<<aggBlocks, 256, 0, stream>>>(H, b2, x, out);
  edge_scatter<<<edgeBlocks, 256, 0, stream>>>(H, ei, attr, out);
}
